// AttentionCountingMLB_9758165697304
// MI455X (gfx1250) — matrix operand order repaired and checked by reading, not compiled, not run on hardware
//
#include <hip/hip_runtime.h>
#include <math.h>

// ---------------------------------------------------------------------------
// AttentionCountingMLB for MI455X (gfx1250, wave32, WMMA).
// ~110 GFLOP vs ~150MB HBM -> compute bound. All GEMMs on
// v_wmma_f32_16x16x32_f16 (f16 in, f32 acc); softmax/LN/ssqrt/Counter in f32.
// This revision: 2-deep software pipelining + dual-tile (shared A) per wave
// + global_prefetch of weight tiles, to overlap ds/global loads with wmma.
// ---------------------------------------------------------------------------

typedef _Float16 half_t;
typedef __attribute__((ext_vector_type(16))) _Float16 v16h;
typedef __attribute__((ext_vector_type(8)))  float    v8f;

#define WMMA_F32_F16(a, b, c) \
  __builtin_amdgcn_wmma_f32_16x16x32_f16(false, (a), false, (b), (short)0, (c), false, false)

static constexpr int kN = 1024, kR = 100, kD = 256, kT = 14, kH = 8;
static constexpr int kMM = 600, kMMp = 608, kOBJ = 36;
static constexpr int kRp = 112;              // rows padded to 7*16
static constexpr int kKp = 128;              // attention K-dim padded to 128

// ------------------------- WMMA fragment helpers ---------------------------
// A-matrix 16x32 f16 (ISA 7.12.2): lane L holds row (L&15);
// halves 0..7 -> K = (L>>4)*8 + i ; halves 8..15 -> K = 16 + (L>>4)*8 + i
__device__ inline v16h frag_a(const half_t* tile, int ld, int lane) {
  int lr = lane & 15, hi = lane >> 4;
  const half_t* p = tile + lr * ld + hi * 8;
  v16h f;
#pragma unroll
  for (int i = 0; i < 8; ++i) { f[i] = p[i]; f[8 + i] = p[16 + i]; }
  return f;
}
// B-matrix 32x16 f16, computing X*W^T: B[k][n] = W[n][k] with W row-major.
// lane L holds column n=(L&15); halves 0..7 -> K = (L>>4)*8 + i ; halves 8..15 -> K = 16 + (L>>4)*8 + i
__device__ inline v16h frag_bT(const half_t* Wtile, int ld, int lane) {
  int n = lane & 15, hi = lane >> 4;
  const half_t* p = Wtile + n * ld + hi * 8;
  v16h f;
#pragma unroll
  for (int i = 0; i < 8; ++i) { f[i] = p[i]; f[8 + i] = p[16 + i]; }
  return f;
}
__device__ inline v8f zero8() {
  v8f z;
#pragma unroll
  for (int i = 0; i < 8; ++i) z[i] = 0.f;
  return z;
}

// Dual-tile k-loop, 2-deep software pipeline: A shared by two B tiles.
// Fragments for step kk+1 are loaded BEFORE the wmma pair consuming step kk,
// so ds/global loads overlap the matrix pipe instead of wait-stalling it.
template <int KT>
__device__ inline void wmma_pair(const half_t* pA, int lda,
                                 const half_t* pB0, const half_t* pB1, int ldb,
                                 int lane, v8f& acc0, v8f& acc1) {
  v16h a  = frag_a(pA, lda, lane);
  v16h b0 = frag_bT(pB0, ldb, lane);
  v16h b1 = frag_bT(pB1, ldb, lane);
#pragma unroll
  for (int kk = 0; kk < KT; ++kk) {
    v16h an, b0n, b1n;
    if (kk + 1 < KT) {
      an  = frag_a(pA + (kk + 1) * 32, lda, lane);
      b0n = frag_bT(pB0 + (kk + 1) * 32, ldb, lane);
      b1n = frag_bT(pB1 + (kk + 1) * 32, ldb, lane);
    }
    acc0 = WMMA_F32_F16(a, b0, acc0);
    acc1 = WMMA_F32_F16(a, b1, acc1);
    if (kk + 1 < KT) { a = an; b0 = b0n; b1 = b1n; }
  }
}

// --------------------------- prep kernels ----------------------------------
__global__ __launch_bounds__(256) void k_combine(
    const float* in_w, const float* in_b, const float* Wq, const float* bq,
    const float* Wk, const float* bk, const float* Wv, const float* bv,
    half_t* Wc, float* bc) {
  int idx = blockIdx.x * 256 + threadIdx.x;
  if (idx >= 3 * 256 * 256) return;
  int m = idx >> 16, rem = idx & 65535, o = rem >> 8, i = rem & 255;
  const float* W = (m == 0) ? Wq : ((m == 1) ? Wk : Wv);
  const float* b = (m == 0) ? bq : ((m == 1) ? bk : bv);
  const float* wrow = in_w + (m * 256 + o) * 256;
  float acc = 0.f;
  for (int t = 0; t < 256; ++t) acc += wrow[t] * W[t * 256 + i];
  Wc[idx] = (half_t)acc;
  if (i == 0) {
    float bb = in_b[m * 256 + o];
    for (int t = 0; t < 256; ++t) bb += wrow[t] * b[t];
    bc[m * 256 + o] = bb;
  }
}

__global__ __launch_bounds__(256) void k_cvt(const float* out_w, const float* l0_w,
                                             half_t* out_wh, half_t* l0_wh) {
  int idx = blockIdx.x * 256 + threadIdx.x;
  if (idx < 65536) { out_wh[idx] = (half_t)out_w[idx]; return; }
  idx -= 65536;
  if (idx < kMMp * 256) {
    int o = idx >> 8, i = idx & 255;
    l0_wh[idx] = (o < kMM) ? (half_t)l0_w[o * 256 + i] : (half_t)0.f;
  }
}

__global__ void k_tables(const float* pw, float* gw, float* gcs) {
  int k = threadIdx.x;
  if (k >= 8) return;
  float w[17]; float sum = 0.f;
  for (int i = 0; i < 17; ++i) { w[i] = (i == 0) ? 0.f : fabsf(pw[k * 17 + i]); sum += w[i]; }
  float inv = 1.f / sum, cs = 0.f;
  for (int i = 0; i < 17; ++i) { w[i] *= inv; cs += w[i]; gw[k * 17 + i] = w[i]; gcs[k * 17 + i] = cs; }
}

__global__ __launch_bounds__(256) void k_qpool(const float* q_emb, float* qpool) {
  int idx = blockIdx.x * 256 + threadIdx.x;
  if (idx >= kN * kD) return;
  int n = idx >> 8, d = idx & 255;
  const float* p = q_emb + n * kT * kD + d;
  float s = 0.f;
  for (int t = 0; t < kT; ++t) s += p[t * kD];
  qpool[idx] = s * (1.f / (float)kT);
}

__global__ __launch_bounds__(256) void k_x1(const float* qpool, const float* l1_w,
                                            const float* l1_b, float* x1) {
  int idx = blockIdx.x * 256 + threadIdx.x;
  if (idx >= kN * kMMp) return;
  int n = idx / kMMp, j = idx % kMMp;
  if (j >= kMM) { x1[idx] = 0.f; return; }
  const float* qp = qpool + n * 256;
  const float* wr = l1_w + j * 256;
  float s = l1_b[j];
  for (int i = 0; i < 256; ++i) s += qp[i] * wr[i];
  x1[idx] = s;
}

// ----------------------- fused attention + MLB kernel ----------------------
static constexpr size_t OFF_A   = 0;        // 112x256 f16 (v_emb, later AO)
static constexpr size_t OFF_Q   = 57344;    // 112x256 f16
static constexpr size_t OFF_K   = 114688;   // 112x256 f16
static constexpr size_t OFF_VT  = 172032;   // 256x128 f16 (V transposed)
static constexpr size_t OFF_S   = 237568;   // 112x112 f32 logits
static constexpr size_t OFF_P   = 287744;   // 112x128 f16 probs
static constexpr size_t OFF_NUM = 316416;   // 112 f32
static constexpr size_t OFF_DEN = 316864;   // 112 f32
static constexpr size_t SMEM_BYTES = 317312;
static constexpr size_t OFF_INTER = OFF_Q;  // 112x256 f32 over sQ+sK
static constexpr size_t OFF_IH    = OFF_VT; // 112x256 f16 over sVt

__global__ __launch_bounds__(256) void k_attn_fused(
    const float* v_emb, const half_t* Wc, const float* bc,
    const half_t* out_wh, const float* out_b, const float* ln_g, const float* ln_b,
    const half_t* l0_wh, const float* l0_b, const float* x1g,
    const float* lo_w, const float* lo_b, float* scores) {
  extern __shared__ char smem[];
  half_t* sA  = (half_t*)(smem + OFF_A);
  half_t* sQ  = (half_t*)(smem + OFF_Q);
  half_t* sK  = (half_t*)(smem + OFF_K);
  half_t* sVt = (half_t*)(smem + OFF_VT);
  float*  sS  = (float*)(smem + OFF_S);
  half_t* sP  = (half_t*)(smem + OFF_P);
  float*  sNum = (float*)(smem + OFF_NUM);
  float*  sDen = (float*)(smem + OFF_DEN);
  float*  sInter = (float*)(smem + OFF_INTER);
  half_t* sIH = (half_t*)(smem + OFF_IH);

  const int n = blockIdx.x, tid = threadIdx.x;
  const int lane = tid & 31, wv = tid >> 5;
  const int nl = lane & 15, mb = (lane >> 4) * 8;

  // Stage 0: v_emb -> f16 LDS (zero-padded rows), zero sVt (padded K rows)
  const float* vin = v_emb + (size_t)n * kR * kD;
  for (int idx = tid; idx < kRp * kD; idx += 256) {
    int r = idx >> 8, c = idx & 255;
    sA[idx] = (r < kR) ? (half_t)vin[r * kD + c] : (half_t)0.f;
  }
  for (int idx = tid; idx < kD * kKp; idx += 256) sVt[idx] = (half_t)0.f;
  __syncthreads();

  // Stage 1: QKV. 3 mats x 7 row-tiles x 8 col-pairs = 168 dual-tiles (K=256)
  for (int t = wv; t < 168; t += 8) {
    int m = t / 56, u = t % 56, rt = u / 8, cp = u % 8;
    const half_t* Wm = Wc + m * 65536;
    if (t + 8 < 168) {                      // global_prefetch_b8 next tiles
      int t2 = t + 8, m2 = t2 / 56, u2 = t2 % 56, cp2 = u2 % 8;
      __builtin_prefetch(Wc + m2 * 65536 + (cp2 * 32) * 256, 0, 0);
    }
    v8f acc0 = zero8(), acc1 = zero8();
    wmma_pair<8>(sA + rt * 16 * kD, kD,
                 Wm + (cp * 32) * 256, Wm + (cp * 32 + 16) * 256, 256,
                 lane, acc0, acc1);
#pragma unroll
    for (int hlf = 0; hlf < 2; ++hlf) {
      const v8f& acc = hlf ? acc1 : acc0;
      int ncol = cp * 32 + hlf * 16 + nl;
      float bias = bc[m * 256 + ncol];
      if (m == 2) {                         // V stored transposed: [feat][key]
        half_t* dst = sVt + ncol * kKp + rt * 16 + mb;
#pragma unroll
        for (int v = 0; v < 8; ++v) dst[v] = (half_t)(acc[v] + bias);
      } else {
        half_t* M = (m == 0) ? sQ : sK;
#pragma unroll
        for (int v = 0; v < 8; ++v) M[(rt * 16 + mb + v) * kD + ncol] = (half_t)(acc[v] + bias);
      }
    }
  }
  __syncthreads();

  // Stage 2: attention, heads sequential
  const float isq = 0.17677669529663687f;  // 1/sqrt(32)
  for (int h = 0; h < kH; ++h) {
    for (int t = wv; t < 49; t += 8) {     // logits: 7x7 tiles, K=32 (1 wmma)
      int rt = t / 7, ck = t % 7;
      v16h a = frag_a(sQ + rt * 16 * kD + h * 32, kD, lane);
      v16h b = frag_bT(sK + ck * 16 * kD + h * 32, kD, lane);
      v8f acc = zero8();
      acc = WMMA_F32_F16(a, b, acc);
      int nn = ck * 16 + nl;
#pragma unroll
      for (int v = 0; v < 8; ++v) sS[(rt * 16 + mb + v) * kRp + nn] = acc[v] * isq;
    }
    __syncthreads();
    if (tid < kRp) {                       // fp32 softmax, rows 100..111 -> 0
      half_t* prow = sP + tid * kKp;
      if (tid < kR) {
        const float* row = sS + tid * kRp;
        float mx = -1e30f;
        for (int c = 0; c < kR; ++c) mx = fmaxf(mx, row[c]);
        float sum = 0.f;
        for (int c = 0; c < kR; ++c) sum += __expf(row[c] - mx);
        float inv = 1.f / sum;
        for (int c = 0; c < kR; ++c) prow[c] = (half_t)(__expf(row[c] - mx) * inv);
        for (int c = kR; c < kKp; ++c) prow[c] = (half_t)0.f;
      } else {
        for (int c = 0; c < kKp; ++c) prow[c] = (half_t)0.f;
      }
    }
    __syncthreads();
    for (int t = wv; t < 7; t += 8) {      // attn @ V: 7 dual-tiles, K=128
      int rt = t;
      v8f acc0 = zero8(), acc1 = zero8();
      wmma_pair<4>(sP + rt * 16 * kKp, kKp,
                   sVt + (h * 32) * kKp, sVt + (h * 32 + 16) * kKp, kKp,
                   lane, acc0, acc1);
#pragma unroll
      for (int v = 0; v < 8; ++v) {
        sA[(rt * 16 + mb + v) * kD + h * 32 + nl] = (half_t)acc0[v];
        sA[(rt * 16 + mb + v) * kD + h * 32 + 16 + nl] = (half_t)acc1[v];
      }
    }
    __syncthreads();
  }

  // Stage 3: out-projection (AO in sA) -> sInter (f32). 7x8 = 56 dual-tiles
  for (int t = wv; t < 56; t += 8) {
    int rt = t / 8, cp = t % 8;
    if (t + 8 < 56) __builtin_prefetch(out_wh + (((t + 8) % 8) * 32) * 256, 0, 0);
    v8f acc0 = zero8(), acc1 = zero8();
    wmma_pair<8>(sA + rt * 16 * kD, kD,
                 out_wh + (cp * 32) * 256, out_wh + (cp * 32 + 16) * 256, 256,
                 lane, acc0, acc1);
#pragma unroll
    for (int hlf = 0; hlf < 2; ++hlf) {
      const v8f& acc = hlf ? acc1 : acc0;
      int ncol = cp * 32 + hlf * 16 + nl;
      float bias = out_b[ncol];
#pragma unroll
      for (int v = 0; v < 8; ++v) sInter[(rt * 16 + mb + v) * kD + ncol] = acc[v] + bias;
    }
  }
  __syncthreads();

  // Stage 4: layernorm -> f16 (sIH), zero pad rows; zero score accumulators
  if (tid < kRp) {
    half_t* drow = sIH + tid * kD;
    if (tid < kR) {
      const float* row = sInter + tid * kD;
      float mu = 0.f;
      for (int c = 0; c < kD; ++c) mu += row[c];
      mu *= (1.f / kD);
      float var = 0.f;
      for (int c = 0; c < kD; ++c) { float d = row[c] - mu; var += d * d; }
      var *= (1.f / kD);
      float inv = rsqrtf(var + 1e-5f);
      for (int c = 0; c < kD; ++c)
        drow[c] = (half_t)(((row[c] - mu) * inv) * ln_g[c] + ln_b[c]);
    } else {
      for (int c = 0; c < kD; ++c) drow[c] = (half_t)0.f;
    }
    sNum[tid] = 0.f; sDen[tid] = 0.f;
  }
  __syncthreads();

  // Stage 5: x0 = inter @ l0_w^T fused signed-sqrt + norm + sigmoid(dot lo_w)
  // 7 row-tiles x 19 col-pairs = 133 dual-tiles
  const float* x1n = x1g + n * kMMp;
  for (int t = wv; t < 133; t += 8) {
    int rt = t / 19, cp = t % 19;
    if (t + 8 < 133) __builtin_prefetch(l0_wh + (((t + 8) % 19) * 32) * 256, 0, 0);
    v8f acc0 = zero8(), acc1 = zero8();
    wmma_pair<8>(sIH + rt * 16 * kD, kD,
                 l0_wh + (cp * 32) * 256, l0_wh + (cp * 32 + 16) * 256, 256,
                 lane, acc0, acc1);
#pragma unroll
    for (int hlf = 0; hlf < 2; ++hlf) {
      const v8f& acc = hlf ? acc1 : acc0;
      int col = cp * 32 + hlf * 16 + nl;
      float b0 = (col < kMM) ? l0_b[col] : 0.f;
      float xv = x1n[col];
      float lw = (col < kMM) ? lo_w[col] : 0.f;
#pragma unroll
      for (int v = 0; v < 8; ++v) {
        float val = (acc[v] + b0) * xv;
        float s = copysignf(sqrtf(fabsf(val)), val);
        int r = rt * 16 + mb + v;
        atomicAdd(&sNum[r], s * lw);       // ds_add_f32
        atomicAdd(&sDen[r], fabsf(val));   // sum(zf^2) = sum|x0*x1|
      }
    }
  }
  __syncthreads();
  if (tid < kR) {
    float nrm = fmaxf(sqrtf(sDen[tid]), 1e-12f);
    float logit = sNum[tid] / nrm + lo_b[0];
    scores[n * kR + tid] = 1.f / (1.f + __expf(-logit));
  }
}

// ------------------------------ Counter ------------------------------------
__device__ inline float plinK(const float* sw, const float* scs, int k, float x) {
  const float* w = sw + k * 17;
  const float* cs = scs + k * 17;
  float y = 16.f * fminf(fmaxf(x, 0.f), 1.f);
  float fl = floorf(y);
  int i = (int)fl; i = min(max(i, 0), 16);
  float f = y - fl;
  int i1 = min(i + 1, 16);
  return cs[i] + f * w[i1];
}

__global__ __launch_bounds__(256) void k_counter(
    const float* scores, const float* boxes, const float* gw, const float* gcs,
    float* pred, float* confOut) {
  __shared__ float s_w[136], s_cs[136];
  __shared__ float s_sc[kR];
  __shared__ float s_att[kOBJ];
  __shared__ float s_box[kOBJ * 4];
  __shared__ float s_area[kOBJ];
  __shared__ float s_dist[kOBJ * kOBJ];
  __shared__ float s_dsc[kOBJ * kOBJ];
  __shared__ float s_sco[kOBJ * kOBJ];
  __shared__ float s_sim[kOBJ * kOBJ];
  __shared__ float s_rs[kOBJ];
  __shared__ float s_red[3];

  const int n = blockIdx.x, tid = threadIdx.x;
  if (tid < 136) { s_w[tid] = gw[tid]; s_cs[tid] = gcs[tid]; }
  if (tid < kR) s_sc[tid] = scores[n * kR + tid];
  if (tid < 3) s_red[tid] = 0.f;
  __syncthreads();

  // top-36 via rank (matches jax top_k descending order, stable ties)
  if (tid < kR) {
    float v = s_sc[tid]; int rk = 0;
    for (int j = 0; j < kR; ++j) {
      float u = s_sc[j];
      rk += (u > v) || (u == v && j < tid);
    }
    if (rk < kOBJ) {
      s_att[rk] = v;
#pragma unroll
      for (int d = 0; d < 4; ++d) s_box[rk * 4 + d] = boxes[((size_t)n * kR + tid) * 4 + d];
    }
  }
  __syncthreads();
  if (tid < kOBJ) {
    float x1 = s_box[tid * 4], y1 = s_box[tid * 4 + 1];
    float x2 = s_box[tid * 4 + 2], y2 = s_box[tid * 4 + 3];
    s_area[tid] = fmaxf(x2 - x1, 0.f) * fmaxf(y2 - y1, 0.f);
  }
  __syncthreads();
  for (int p = tid; p < kOBJ * kOBJ; p += 256) {
    int i = p / kOBJ, j = p % kOBJ;
    float ltx = fmaxf(s_box[i * 4], s_box[j * 4]);
    float lty = fmaxf(s_box[i * 4 + 1], s_box[j * 4 + 1]);
    float rbx = fminf(s_box[i * 4 + 2], s_box[j * 4 + 2]);
    float rby = fminf(s_box[i * 4 + 3], s_box[j * 4 + 3]);
    float iw = fmaxf(rbx - ltx, 0.f), ih = fmaxf(rby - lty, 0.f);
    float inter = iw * ih;
    float iou = inter / (s_area[i] + s_area[j] - inter + 1e-12f);
    float dist = 1.f - iou;
    s_dist[p] = dist;
    float rel = s_att[i] * s_att[j];
    s_sco[p] = plinK(s_w, s_cs, 0, rel) * plinK(s_w, s_cs, 1, dist);
    s_dsc[p] = plinK(s_w, s_cs, 3, rel) * plinK(s_w, s_cs, 4, dist);
  }
  __syncthreads();
  for (int p = tid; p < kOBJ * kOBJ; p += 256) {
    int b = p / kOBJ, c = p % kOBJ;
    float sim = plinK(s_w, s_cs, 2, 1.f - fabsf(s_att[b] - s_att[c]));
    for (int a = 0; a < kOBJ; ++a)
      sim *= plinK(s_w, s_cs, 2, 1.f - fabsf(s_dsc[a * kOBJ + b] - s_dsc[a * kOBJ + c]));
    s_sim[p] = sim;
  }
  __syncthreads();
  if (tid < kOBJ) {
    float s = 0.f;
    for (int c = 0; c < kOBJ; ++c) s += s_sim[tid * kOBJ + c];
    s_rs[tid] = s;
  }
  __syncthreads();
  float tpart = 0.f, dpart = 0.f, apart = 0.f;
  for (int p = tid; p < kOBJ * kOBJ; p += 256) {
    int b = p / kOBJ, c = p % kOBJ;
    tpart += s_sco[p] / (s_rs[b] * s_rs[c]);
    dpart += fabsf(plinK(s_w, s_cs, 6, s_dist[p]) - 0.5f);
  }
  if (tid < kOBJ) {
    tpart += plinK(s_w, s_cs, 0, s_att[tid] * s_att[tid]) / s_rs[tid];
    apart = fabsf(plinK(s_w, s_cs, 5, s_att[tid]) - 0.5f);
  }
  atomicAdd(&s_red[0], tpart);
  atomicAdd(&s_red[1], apart);
  atomicAdd(&s_red[2], dpart);
  __syncthreads();
  if (tid == 0) {
    float total = sqrtf(s_red[0] + 1e-20f);
    float conf = plinK(s_w, s_cs, 7, s_red[1] / 36.f + s_red[2] / 1296.f);
    confOut[n] = conf;
    s_red[0] = total; s_red[1] = conf;
  }
  __syncthreads();
  if (tid < kOBJ + 1) {
    float total = s_red[0], conf = s_red[1];
    float s = fminf(fmaxf(total, 0.f), 36.f);
    float fl = floorf(s);
    int i = (int)fl; i = min(max(i, 0), 36);
    float f = s - fl;
    int i1 = min(i + 1, 36);
    float oh = (1.f - f) * ((tid == i) ? 1.f : 0.f) + f * ((tid == i1) ? 1.f : 0.f);
    pred[n * (kOBJ + 1) + tid] = oh * conf;
  }
}

// ------------------------------ launcher -----------------------------------
extern "C" void kernel_launch(void* const* d_in, const int* in_sizes, int n_in,
                              void* d_out, int out_size, void* d_ws, size_t ws_size,
                              hipStream_t stream) {
  (void)in_sizes; (void)n_in; (void)out_size; (void)ws_size;
  const float* v_emb = (const float*)d_in[0];
  const float* q_emb = (const float*)d_in[1];
  const float* boxes = (const float*)d_in[2];
  const float* Wq = (const float*)d_in[3];  const float* bq = (const float*)d_in[4];
  const float* Wk = (const float*)d_in[5];  const float* bk = (const float*)d_in[6];
  const float* Wv = (const float*)d_in[7];  const float* bv = (const float*)d_in[8];
  const float* in_w = (const float*)d_in[9];  const float* in_b = (const float*)d_in[10];
  const float* out_w = (const float*)d_in[11]; const float* out_b = (const float*)d_in[12];
  const float* ln_g = (const float*)d_in[13];  const float* ln_b = (const float*)d_in[14];
  const float* l0_w = (const float*)d_in[15];  const float* l0_b = (const float*)d_in[16];
  const float* l1_w = (const float*)d_in[17];  const float* l1_b = (const float*)d_in[18];
  const float* lo_w = (const float*)d_in[19];  const float* lo_b = (const float*)d_in[20];
  const float* pw = (const float*)d_in[21];

  char* ws = (char*)d_ws;
  half_t* Wc     = (half_t*)(ws + 0);         // 3*256*256 f16 = 393216 B
  half_t* out_wh = (half_t*)(ws + 393216);    // 65536 f16    = 131072 B
  half_t* l0_wh  = (half_t*)(ws + 524288);    // 608*256 f16  = 311296 B
  float*  bc     = (float*)(ws + 835584);     // 768 f32
  float*  gw     = (float*)(ws + 838656);     // 136 f32
  float*  gcs    = (float*)(ws + 839296);     // 136 f32
  float*  qpool  = (float*)(ws + 839936);     // 1024*256 f32 = 1048576 B
  float*  x1     = (float*)(ws + 1888512);    // 1024*608 f32 = 2490368 B  (end ~4.38MB)

  float* scores = (float*)d_out;                       // N*R
  float* pred   = scores + kN * kR;                    // N*37
  float* conf   = pred + kN * (kOBJ + 1);              // N

  k_combine<<<768, 256, 0, stream>>>(in_w, in_b, Wq, bq, Wk, bk, Wv, bv, Wc, bc);
  k_cvt<<<864, 256, 0, stream>>>(out_w, l0_w, out_wh, l0_wh);
  k_tables<<<1, 32, 0, stream>>>(pw, gw, gcs);
  k_qpool<<<1024, 256, 0, stream>>>(q_emb, qpool);
  k_x1<<<2432, 256, 0, stream>>>(qpool, l1_w, l1_b, x1);
  k_attn_fused<<<kN, 256, SMEM_BYTES, stream>>>(
      v_emb, Wc, bc, out_wh, out_b, ln_g, ln_b, l0_wh, l0_b, x1, lo_w, lo_b, scores);
  k_counter<<<kN, 256, 0, stream>>>(scores, boxes, gw, gcs, pred, conf);
}
